// Decoder_60387240182010
// MI455X (gfx1250) — matrix operand order repaired and checked by reading, not compiled, not run on hardware
//
#include <hip/hip_runtime.h>
#include <math.h>

// ---------------- CDNA5 WMMA types ----------------
typedef __attribute__((ext_vector_type(16))) __bf16 v16bf;
typedef __attribute__((ext_vector_type(8)))  __bf16 bf16x8;
typedef __attribute__((ext_vector_type(8)))  float  v8f;

// Problem dims (fixed by the reference setup)
constexpr int Bn = 64;    // batch
constexpr int Ln = 256;   // latent
constexpr int Hn = 1024;  // hidden
constexpr int On = 512;   // output
constexpr int Sn = 256;   // seq_len
constexpr int KT = Hn / 32;  // 32 k-tiles of depth 32 over H

// ---------------------------------------------------------------------------
// Pack a row-major f32 matrix W[N][K] (optionally W = src + src2) into the
// per-lane bf16 B-fragment layout for v_wmma_f32_16x16x32_bf16:
//   fragment (nt, kt): lane l holds 16 contiguous bf16 with
//     n = nt*16 + (l & 15),  k = kt*32 + (l>=16 ? 16 : 0) + e,  e = 0..15
// Stored contiguously: dst[(((nt*KT)+kt)*32 + lane)*16 + e]
// ---------------------------------------------------------------------------
__global__ void pack_b_kernel(const float* __restrict__ src,
                              const float* __restrict__ src2,
                              __bf16* __restrict__ dst,
                              int N, int K) {
    int ktiles = K / 32;
    int total  = (N / 16) * ktiles * 32;           // lane-slots
    int id = blockIdx.x * blockDim.x + threadIdx.x;
    if (id >= total) return;
    int lane = id & 31;
    int kt   = (id >> 5) % ktiles;
    int nt   = id / (32 * ktiles);
    int n    = nt * 16 + (lane & 15);
    int kb   = kt * 32 + ((lane >> 4) & 1) * 16;
    const float* s  = src + (size_t)n * K + kb;
    const float* s2 = src2 ? (src2 + (size_t)n * K + kb) : nullptr;
    __bf16* d = dst + (size_t)id * 16;
#pragma unroll
    for (int e = 0; e < 16; ++e) {
        float v = s[e];
        if (s2) v += s2[e];
        d[e] = (__bf16)v;
    }
}

__global__ void bias_kernel(const float* __restrict__ bih,
                            const float* __restrict__ bhh,
                            float* __restrict__ bias) {
    int i = blockIdx.x * blockDim.x + threadIdx.x;
    if (i < 4 * Hn) bias[i] = bih[i] + bhh[i];
}

// h0 = latent @ fc_w.T + fc_b ; c = 0 ; h stored as bf16
__global__ void h0_kernel(const float* __restrict__ latent,
                          const float* __restrict__ fc_w,
                          const float* __restrict__ fc_b,
                          __bf16* __restrict__ h_bf,
                          float* __restrict__ c_state) {
    int id = blockIdx.x * blockDim.x + threadIdx.x;   // 0 .. B*H
    if (id >= Bn * Hn) return;
    int b = id / Hn, j = id % Hn;
    float acc = fc_b[j];
    const float* lr = latent + (size_t)b * Ln;
    const float* wr = fc_w + (size_t)j * Ln;
#pragma unroll 4
    for (int k = 0; k < Ln; ++k) acc = fmaf(lr[k], wr[k], acc);
    h_bf[id]    = (__bf16)acc;
    c_state[id] = 0.0f;
}

// ---------------- fragment loaders ----------------
// A-fragment: 16x32 bf16 from row-major h[B][H].
// 16-bit A layout (matches packed B): lane m (0-15): K 0..15
//                                     lane m+16   : K 16..31
__device__ __forceinline__ v16bf load_a_frag(const __bf16* __restrict__ h_bf,
                                             int mt, int kt, int lane) {
    int m  = mt * 16 + (lane & 15);
    int k0 = kt * 32 + ((lane >> 4) & 1) * 16;
    const __bf16* p = h_bf + (size_t)m * Hn + k0;
    bf16x8 lo = *(const bf16x8*)(p);        // K = k0..k0+7
    bf16x8 hi = *(const bf16x8*)(p + 8);    // K = k0+8..k0+15
    v16bf a;
#pragma unroll
    for (int i = 0; i < 8; ++i) { a[i] = lo[i]; a[8 + i] = hi[i]; }
    return a;
}

__device__ __forceinline__ v16bf load_b_frag(const __bf16* __restrict__ packed,
                                             int nt, int kt, int ktiles, int lane) {
    const __bf16* p = packed + ((((size_t)nt * ktiles) + kt) * 32 + lane) * 16;
    return *(const v16bf*)p;   // 32B contiguous, 32B aligned
}

__device__ __forceinline__ float sigmoidf_(float x) {
    return 1.0f / (1.0f + __expf(-x));
}
// tanh(x) = 1 - 2/(exp(2x)+1); single v_exp TRANS op, saturates correctly:
//   x >> 0: exp->inf -> 1 ; x << 0: exp->0 -> -1
__device__ __forceinline__ float fast_tanhf_(float x) {
    return 1.0f - 2.0f / (__expf(2.0f * x) + 1.0f);
}

// ------------------------- device-side role bodies -------------------------
// Recurrence: gates = h @ W.T + bias ; pointwise ; h_new -> h_out.
// One block per 16-column j-slice; 512 threads = 16 waves = {gate}x{m-tile}.
__device__ __forceinline__ void lstm_step_body(
        const __bf16* __restrict__ h_bf, const __bf16* __restrict__ Wpacked,
        const float* __restrict__ bias, float* __restrict__ c_state,
        __bf16* __restrict__ h_out, int jt) {
    __shared__ float gl[4][Bn][16];   // 16 KB: i/f/g/o tiles for this j-slice

    const int tid  = threadIdx.x;
    const int lane = tid & 31;
    const int w    = tid >> 5;     // 0..15
    const int g    = w >> 2;       // gate index
    const int mt   = w & 3;        // batch m-tile
    const int nt   = g * (Hn / 16) + jt;   // n-tile in 4H dimension

    v8f acc = {};
#pragma unroll 4
    for (int kt = 0; kt < KT; ++kt) {
        v16bf a = load_a_frag(h_bf, mt, kt, lane);
        v16bf b = load_b_frag(Wpacked, nt, kt, KT, lane);
        acc = __builtin_amdgcn_wmma_f32_16x16x32_bf16(
                  false, a, false, b, (short)0, acc, false, false);
    }

    // C layout: lanes 0-15 -> M = r, lanes 16-31 -> M = 8 + r ; N = lane & 15
    const int mrow = mt * 16 + ((lane >> 4) & 1) * 8;
    const int col  = lane & 15;
#pragma unroll
    for (int r = 0; r < 8; ++r) gl[g][mrow + r][col] = acc[r];
    __syncthreads();

    // fused pointwise: 64 rows x 16 cols = 1024 elems, 512 threads -> 2 each
#pragma unroll
    for (int it = 0; it < 2; ++it) {
        int idx = tid + it * 512;
        int m   = idx >> 4;
        int jc  = idx & 15;
        int j   = jt * 16 + jc;
        float xi = gl[0][m][jc] + bias[0 * Hn + j];
        float xf = gl[1][m][jc] + bias[1 * Hn + j];
        float xg = gl[2][m][jc] + bias[2 * Hn + j];
        float xo = gl[3][m][jc] + bias[3 * Hn + j];
        float ig = sigmoidf_(xi);
        float fg = sigmoidf_(xf);
        float gg = fast_tanhf_(xg);
        float og = sigmoidf_(xo);
        size_t ci = (size_t)m * Hn + j;
        float cn  = fg * c_state[ci] + ig * gg;
        c_state[ci] = cn;
        float hn  = og * fast_tanhf_(cn);
        h_out[ci] = (__bf16)hn;
    }
}

// Projection: out[:, t, :] = h @ out_w.T + out_b. One 16x16 tile per wave.
__device__ __forceinline__ void proj_body(
        const __bf16* __restrict__ h_bf, const __bf16* __restrict__ OWpacked,
        const float* __restrict__ out_b, float* __restrict__ out,
        int t, int tile) {
    const int lane = threadIdx.x & 31;
    const int mt   = tile >> 5;    // 0..3
    const int ot   = tile & 31;    // 0..31

    v8f acc = {};
#pragma unroll 4
    for (int kt = 0; kt < KT; ++kt) {
        v16bf a = load_a_frag(h_bf, mt, kt, lane);
        v16bf b = load_b_frag(OWpacked, ot, kt, KT, lane);
        acc = __builtin_amdgcn_wmma_f32_16x16x32_bf16(
                  false, a, false, b, (short)0, acc, false, false);
    }

    const int o     = ot * 16 + (lane & 15);
    const int mbase = mt * 16 + ((lane >> 4) & 1) * 8;
    const float ob  = out_b[o];
#pragma unroll
    for (int r = 0; r < 8; ++r) {
        int m = mbase + r;
        out[(size_t)m * Sn * On + (size_t)t * On + o] = acc[r] + ob;
    }
}

// ---------------------------- kernels ----------------------------
// Standalone recurrence (step 0). Grid: 64 blocks x 512 threads.
__global__ void __launch_bounds__(512)
lstm_step_kernel(const __bf16* __restrict__ h_bf,
                 const __bf16* __restrict__ Wpacked,
                 const float*  __restrict__ bias,
                 float*        __restrict__ c_state,
                 __bf16*       __restrict__ h_out) {
    lstm_step_body(h_bf, Wpacked, bias, c_state, h_out, blockIdx.x);
}

// Standalone projection (last step). Grid: 8 blocks x 512 threads (16 waves).
__global__ void __launch_bounds__(512)
proj_kernel(const __bf16* __restrict__ h_bf,
            const __bf16* __restrict__ OWpacked,
            const float*  __restrict__ out_b,
            float*        __restrict__ out,
            int t) {
    int tile = blockIdx.x * 16 + (threadIdx.x >> 5);   // 0..127
    proj_body(h_bf, OWpacked, out_b, out, t, tile);
}

// Fused: blocks 0..63 run recurrence step t (h_cur -> h_next);
//        blocks 64..71 run projection for step t-1 (reads h_cur).
// Both roles only read h_cur; no cross-role dependency -> they overlap.
__global__ void __launch_bounds__(512)
fused_step_kernel(const __bf16* __restrict__ h_cur,
                  const __bf16* __restrict__ Wpacked,
                  const float*  __restrict__ bias,
                  float*        __restrict__ c_state,
                  __bf16*       __restrict__ h_next,
                  const __bf16* __restrict__ OWpacked,
                  const float*  __restrict__ out_b,
                  float*        __restrict__ out,
                  int t_prev) {
    if (blockIdx.x < (unsigned)(Hn / 16)) {
        lstm_step_body(h_cur, Wpacked, bias, c_state, h_next, blockIdx.x);
    } else {
        int tile = (blockIdx.x - Hn / 16) * 16 + (threadIdx.x >> 5); // 0..127
        proj_body(h_cur, OWpacked, out_b, out, t_prev, tile);
    }
}

// ---------------------------------------------------------------------------
extern "C" void kernel_launch(void* const* d_in, const int* in_sizes, int n_in,
                              void* d_out, int out_size, void* d_ws, size_t ws_size,
                              hipStream_t stream) {
    const float* latent = (const float*)d_in[0];
    // d_in[1] = seq_len scalar (fixed at 256 by setup)
    const float* fc_w   = (const float*)d_in[2];
    const float* fc_b   = (const float*)d_in[3];
    const float* w_ih   = (const float*)d_in[4];
    const float* w_hh   = (const float*)d_in[5];
    const float* b_ih   = (const float*)d_in[6];
    const float* b_hh   = (const float*)d_in[7];
    const float* out_w  = (const float*)d_in[8];
    const float* out_b  = (const float*)d_in[9];
    float* out = (float*)d_out;

    // workspace carving (~17.6 MB total)
    char* ws = (char*)d_ws;
    size_t off = 0;
    auto carve = [&](size_t bytes) -> void* {
        void* p = ws + off;
        off = (off + bytes + 255) & ~(size_t)255;
        return p;
    };
    __bf16* Wc   = (__bf16*)carve((size_t)4 * Hn * Hn * 2);  // w_ih + w_hh, packed
    __bf16* Whh  = (__bf16*)carve((size_t)4 * Hn * Hn * 2);  // w_hh, packed (step 0)
    __bf16* OW   = (__bf16*)carve((size_t)On * Hn * 2);      // out_w, packed
    float*  bias = (float*) carve((size_t)4 * Hn * 4);
    float*  cst  = (float*) carve((size_t)Bn * Hn * 4);
    __bf16* hA   = (__bf16*)carve((size_t)Bn * Hn * 2);      // ping
    __bf16* hB   = (__bf16*)carve((size_t)Bn * Hn * 2);      // pong

    // --- prep ---
    int slots_w = (4 * Hn / 16) * (Hn / 32) * 32;   // 262144
    int slots_o = (On / 16) * (Hn / 32) * 32;       // 32768
    pack_b_kernel<<<(slots_w + 255) / 256, 256, 0, stream>>>(w_ih, w_hh, Wc, 4 * Hn, Hn);
    pack_b_kernel<<<(slots_w + 255) / 256, 256, 0, stream>>>(w_hh, nullptr, Whh, 4 * Hn, Hn);
    pack_b_kernel<<<(slots_o + 255) / 256, 256, 0, stream>>>(out_w, nullptr, OW, On, Hn);
    bias_kernel<<<(4 * Hn + 255) / 256, 256, 0, stream>>>(b_ih, b_hh, bias);
    h0_kernel<<<(Bn * Hn + 255) / 256, 256, 0, stream>>>(latent, fc_w, fc_b, hA, cst);

    // --- 256 sequential steps, projection fused into the following step ---
    // G_t: h_t -> h_{t+1};  P_t: out_t = proj(h_{t+1}).
    __bf16* hin  = hA;
    __bf16* hout = hB;

    // step 0: x0 == 0, so gates use w_hh only
    lstm_step_kernel<<<Hn / 16, 512, 0, stream>>>(hin, Whh, bias, cst, hout);
    { __bf16* tmp = hin; hin = hout; hout = tmp; }

    for (int t = 1; t < Sn; ++t) {
        // blocks 0..63: G_t (hin -> hout); blocks 64..71: P_{t-1} (reads hin)
        fused_step_kernel<<<Hn / 16 + 8, 512, 0, stream>>>(
            hin, Wc, bias, cst, hout, OW, out_b, out, t - 1);
        __bf16* tmp = hin; hin = hout; hout = tmp;
    }

    // final projection P_{S-1} reads h_S (now in hin after the last swap)
    proj_kernel<<<8, 512, 0, stream>>>(hin, OW, out_b, out, Sn - 1);

    (void)in_sizes; (void)n_in; (void)out_size; (void)ws_size;
}
